// MultiLevel_BlockAttention_52390011077421
// MI455X (gfx1250) — matrix operand order repaired and checked by reading, not compiled, not run on hardware
//
#include <hip/hip_runtime.h>

#define B_ 8
#define S_ 2048
#define E_ 512
#define H_ 4
#define D_ 128

typedef __attribute__((ext_vector_type(16))) __bf16 v16bf;
typedef __attribute__((ext_vector_type(8)))  float  v8f;
typedef __attribute__((ext_vector_type(4)))  float  v4f;

union BF16x16 {
  v16bf v;
  uint4 q[2];
};

__device__ __forceinline__ unsigned short f2bf(float f) {
  unsigned u = __float_as_uint(f);
  u += 0x7FFFu + ((u >> 16) & 1u);   // round-to-nearest-even bf16
  return (unsigned short)(u >> 16);
}

// async memory->LDS copy of 16 bytes; ldsOff is the workgroup-relative LDS byte address
__device__ __forceinline__ void async_b128_to_lds(unsigned ldsOff,
                                                  unsigned long long gaddr) {
  asm volatile("global_load_async_to_lds_b128 %0, %1, off"
               :: "v"(ldsOff), "v"(gaddr) : "memory");
}
__device__ __forceinline__ void wait_async0() {
  asm volatile("s_wait_asynccnt 0" ::: "memory");
}

// ---------------------------------------------------------------- cvt fp32->bf16
__global__ void cvt_f32_bf16(const float* __restrict__ src,
                             unsigned short* __restrict__ dst, int n) {
  int i = blockIdx.x * blockDim.x + threadIdx.x;
  int stride = gridDim.x * blockDim.x;
  for (; i < n; i += stride)
    dst[i] = f2bf(__builtin_nontemporal_load(src + i));
}

// ---------------------------------------------------------------- QKV projection
// grid = (4, 1024, 3), block = 256 (8 waves). Wave computes one 16x16 tile of
// (x @ W[z]) + bias[z].  z: 0=Q, 1=K(store transposed), 2=V.
__global__ void __launch_bounds__(256)
qkv_kernel(const unsigned short* __restrict__ xb,   // [B*S, E] bf16
           const unsigned short* __restrict__ Wb,   // [3, E, E] bf16
           const float* __restrict__ bq,
           const float* __restrict__ bk,
           const float* __restrict__ bv,
           unsigned short* __restrict__ Qb,         // [B,H,S,D] bf16
           unsigned short* __restrict__ Ktb,        // [B,H,D,S] bf16
           unsigned short* __restrict__ Vb) {       // [B,H,S,D] bf16
  __shared__ unsigned short xa[16 * E_];            // 16 KB at LDS offset 0
  const int z   = blockIdx.z;
  const int r0  = blockIdx.y * 16;                  // row block in [0, B*S)
  const int tid = threadIdx.x;

  { // stage 16x512 bf16 rows of x into LDS with async-to-LDS (1024 x b128)
    unsigned long long src = (unsigned long long)(xb + (size_t)r0 * E_);
    for (int i = tid; i < 1024; i += 256)
      async_b128_to_lds((unsigned)(i * 16), src + (unsigned)(i * 16));
    wait_async0();
  }
  __syncthreads();

  const int wave = tid >> 5;
  const int lane = tid & 31;
  const int half = lane >> 4;
  const int mrow = lane & 15;
  const int n    = lane & 15;
  const int n0   = blockIdx.x * 128 + wave * 16;

  const unsigned short* W = Wb + (size_t)z * E_ * E_;
  v8f acc = {};
  for (int kk = 0; kk < 16; ++kk) {
    BF16x16 a, b;
    int abase = mrow * E_ + kk * 32 + half * 8;     // A: VGPR0-3 = K[half*8..+7]
    a.q[0] = *(const uint4*)(xa + abase);
    a.q[1] = *(const uint4*)(xa + abase + 16);      // VGPR4-7 = K[+16..]
    const unsigned short* wrow = W + (size_t)(kk * 32 + half * 8) * E_ + n0 + n; // B: lane=N, elems=K (table order)
    for (int s = 0; s < 8; ++s) {
      b.v[s] = __builtin_bit_cast(__bf16, wrow[(size_t)s * E_]);
      b.v[s + 8] = __builtin_bit_cast(__bf16, wrow[(size_t)(s + 16) * E_]);
    }
    acc = __builtin_amdgcn_wmma_f32_16x16x32_bf16(false, a.v, false, b.v,
                                                  (short)0, acc, false, false);
  }

  // bias: [S,1] broadcast across feature dim -> indexed by sequence row
  const float* bias = (z == 0) ? bq : (z == 1) ? bk : bv;
  const int s0   = (blockIdx.y & (S_ / 16 - 1)) * 16;
  const int bat  = blockIdx.y >> 7;                 // 128 row-blocks per batch
  const int h    = n0 >> 7;
  const int d0   = n0 & (D_ - 1);
  const size_t bh = (size_t)bat * H_ + h;
#pragma unroll
  for (int i = 0; i < 8; ++i) acc[i] += bias[s0 + i + 8 * half];

  if (z == 1) {
    // store K transposed: [bh][d][s]; lane's 8 M-values are consecutive s
    unsigned pk0 = (unsigned)f2bf(acc[0]) | ((unsigned)f2bf(acc[1]) << 16);
    unsigned pk1 = (unsigned)f2bf(acc[2]) | ((unsigned)f2bf(acc[3]) << 16);
    unsigned pk2 = (unsigned)f2bf(acc[4]) | ((unsigned)f2bf(acc[5]) << 16);
    unsigned pk3 = (unsigned)f2bf(acc[6]) | ((unsigned)f2bf(acc[7]) << 16);
    *(uint4*)(Ktb + (bh * D_ + d0 + n) * (size_t)S_ + s0 + 8 * half) =
        make_uint4(pk0, pk1, pk2, pk3);
  } else {
    unsigned short* dst =
        (z == 0 ? Qb : Vb) + (bh * S_ + s0) * (size_t)D_ + d0 + n;
#pragma unroll
    for (int i = 0; i < 8; ++i)
      dst[(size_t)(i + 8 * half) * D_] = f2bf(acc[i]);
  }
}

// ---------------------------------------------------------------- fused attention
// grid = (S/16, B*H), block = 256 (8 waves). Dynamic LDS layout:
#define SMEM_SC   0
#define SMEM_AB   (16 * S_ * 4)                 // 131072: scores fp32 [16][2048]
#define SMEM_QA   (SMEM_AB + 16 * S_ * 2)       // 196608: attn bf16 [16][2048]
#define SMEM_RED  (SMEM_QA + 16 * D_ * 2)       // 200704: q bf16 [16][128]
#define SMEM_TOT  (SMEM_RED + 16 * 16 * 4)      // 201728: red fp32 [16][16]

__global__ void __launch_bounds__(256)
attn_kernel(const unsigned short* __restrict__ Qb,
            const unsigned short* __restrict__ Ktb,
            const unsigned short* __restrict__ Vb,
            const float* __restrict__ Wepi,
            const float* __restrict__ bo,
            float* __restrict__ out_eff,        // [B,S,E]
            float* __restrict__ out_attn) {     // [B*H,S,S]
  extern __shared__ char smem[];
  float*          sc = (float*)(smem + SMEM_SC);
  unsigned short* ab = (unsigned short*)(smem + SMEM_AB);
  unsigned short* qa = (unsigned short*)(smem + SMEM_QA);
  float*          red = (float*)(smem + SMEM_RED);

  const int s0   = blockIdx.x * 16;
  const int bh   = blockIdx.y;
  const int tid  = threadIdx.x;
  const int wave = tid >> 5;
  const int lane = tid & 31;
  const int half = lane >> 4;
  const int mrow = lane & 15;
  const int n    = lane & 15;

  { // stage q block [16][128] bf16 via async-to-LDS (4 KB, one b128/thread)
    unsigned long long src =
        (unsigned long long)(Qb + ((size_t)bh * S_ + s0) * D_);
    async_b128_to_lds((unsigned)(SMEM_QA + tid * 16), src + (unsigned)(tid * 16));
    wait_async0();
  }
  __syncthreads();

  BF16x16 a0[4];
#pragma unroll
  for (int kk = 0; kk < 4; ++kk) {
    int base = mrow * D_ + kk * 32 + half * 8;
    a0[kk].q[0] = *(const uint4*)(qa + base);
    a0[kk].q[1] = *(const uint4*)(qa + base + 16);
  }

  const float scale = 0.022097086912079608f;     // 1/sqrt(S)
  const unsigned short* Kt = Ktb + (size_t)bh * D_ * S_;

  // ---- phase 2: scores = q @ k^T * scale, into LDS
  for (int j = 0; j < 16; ++j) {
    int t0 = (wave * 16 + j) * 16;
    v8f acc = {};
#pragma unroll
    for (int kk = 0; kk < 4; ++kk) {
      BF16x16 bmat;
      const unsigned short* kr = Kt + (size_t)(kk * 32 + half * 8) * S_ + t0 + n;
      for (int s = 0; s < 8; ++s) {
        bmat.v[s] = __builtin_bit_cast(__bf16, kr[(size_t)s * S_]);
        bmat.v[s + 8] = __builtin_bit_cast(__bf16, kr[(size_t)(s + 16) * S_]);
      }
      acc = __builtin_amdgcn_wmma_f32_16x16x32_bf16(false, a0[kk].v, false,
                                                    bmat.v, (short)0, acc,
                                                    false, false);
    }
#pragma unroll
    for (int i = 0; i < 8; ++i)
      sc[(i + 8 * half) * S_ + t0 + n] = acc[i] * scale;
  }
  __syncthreads();

  // ---- phase 3: row softmax, * Wepi, write fp32 attn (NT) + bf16 copy to LDS
  {
    const int r = tid >> 4;
    const int c = tid & 15;
    float* row = sc + r * S_ + c * 128;
    float lmax = -3.4e38f;
    for (int i = 0; i < 128; ++i) lmax = fmaxf(lmax, row[i]);
    red[r * 16 + c] = lmax;
    __syncthreads();
    float rmax = -3.4e38f;
#pragma unroll
    for (int i2 = 0; i2 < 16; ++i2) rmax = fmaxf(rmax, red[r * 16 + i2]);
    __syncthreads();
    float lsum = 0.f;
    for (int i = 0; i < 128; ++i) {
      float e = __expf(row[i] - rmax);
      row[i] = e;
      lsum += e;
    }
    red[r * 16 + c] = lsum;
    __syncthreads();
    float rsum = 0.f;
#pragma unroll
    for (int i2 = 0; i2 < 16; ++i2) rsum += red[r * 16 + i2];
    float inv = 1.0f / rsum;

    const float* we = Wepi + (size_t)(s0 + r) * S_ + c * 128;
    float* oa = out_attn + ((size_t)bh * S_ + (s0 + r)) * S_ + c * 128;
    unsigned short* abr = ab + r * S_ + c * 128;
    for (int i = 0; i < 128; i += 4) {
      v4f wv = *(const v4f*)(we + i);
      v4f o;
      o.x = row[i + 0] * inv * wv.x;
      o.y = row[i + 1] * inv * wv.y;
      o.z = row[i + 2] * inv * wv.z;
      o.w = row[i + 3] * inv * wv.w;
      __builtin_nontemporal_store(o, (v4f*)(oa + i));   // streaming output
      unsigned p0 = (unsigned)f2bf(o.x) | ((unsigned)f2bf(o.y) << 16);
      unsigned p1 = (unsigned)f2bf(o.z) | ((unsigned)f2bf(o.w) << 16);
      *(uint2*)(abr + i) = make_uint2(p0, p1);
    }
  }
  __syncthreads();

  // ---- phase 4: effect = attn @ v + bo, wave handles 16 of D=128 columns
  {
    const unsigned short* V = Vb + (size_t)bh * S_ * D_;
    const int nd0 = wave * 16;
    v8f acc = {};
    for (int tt = 0; tt < 64; ++tt) {
      BF16x16 amat, bmat;
      int base = mrow * S_ + tt * 32 + half * 8;
      amat.q[0] = *(const uint4*)(ab + base);
      amat.q[1] = *(const uint4*)(ab + base + 16);
      const unsigned short* vr = V + (size_t)(tt * 32 + half * 8) * D_ + nd0 + n;
      for (int s = 0; s < 8; ++s) {
        bmat.v[s] = __builtin_bit_cast(__bf16, vr[(size_t)s * D_]);
        bmat.v[s + 8] = __builtin_bit_cast(__bf16, vr[(size_t)(s + 16) * D_]);
      }
      acc = __builtin_amdgcn_wmma_f32_16x16x32_bf16(false, amat.v, false,
                                                    bmat.v, (short)0, acc,
                                                    false, false);
    }
    const int bat = bh >> 2;
    const int h   = bh & 3;
#pragma unroll
    for (int i = 0; i < 8; ++i) {
      int s = s0 + i + 8 * half;
      __builtin_nontemporal_store(
          acc[i] + bo[s],
          &out_eff[((size_t)bat * S_ + s) * E_ + h * D_ + nd0 + n]);
    }
  }
}

// ---------------------------------------------------------------- launcher
extern "C" void kernel_launch(void* const* d_in, const int* in_sizes, int n_in,
                              void* d_out, int out_size, void* d_ws,
                              size_t ws_size, hipStream_t stream) {
  (void)in_sizes; (void)n_in; (void)out_size; (void)ws_size;
  const float* x    = (const float*)d_in[0];
  const float* Wq   = (const float*)d_in[1];
  const float* Wk   = (const float*)d_in[2];
  const float* Wv   = (const float*)d_in[3];
  const float* Wepi = (const float*)d_in[4];
  const float* bq   = (const float*)d_in[5];
  const float* bk   = (const float*)d_in[6];
  const float* bv   = (const float*)d_in[7];
  const float* bo   = (const float*)d_in[8];

  const size_t qkvElems = (size_t)B_ * H_ * S_ * D_;  // == B*S*E
  unsigned short* Qb  = (unsigned short*)d_ws;
  unsigned short* Ktb = Qb + qkvElems;
  unsigned short* Vb  = Ktb + qkvElems;
  unsigned short* xb  = Vb + qkvElems;
  unsigned short* Wb  = xb + qkvElems;                // 3*E*E bf16

  cvt_f32_bf16<<<1024, 256, 0, stream>>>(x, xb, (int)qkvElems);
  cvt_f32_bf16<<<256, 256, 0, stream>>>(Wq, Wb, E_ * E_);
  cvt_f32_bf16<<<256, 256, 0, stream>>>(Wk, Wb + E_ * E_, E_ * E_);
  cvt_f32_bf16<<<256, 256, 0, stream>>>(Wv, Wb + 2 * E_ * E_, E_ * E_);

  qkv_kernel<<<dim3(4, (B_ * S_) / 16, 3), 256, 0, stream>>>(
      xb, Wb, bq, bk, bv, Qb, Ktb, Vb);

  float* out_eff  = (float*)d_out;
  float* out_attn = out_eff + (size_t)B_ * S_ * E_;
  hipFuncSetAttribute((const void*)attn_kernel,
                      hipFuncAttributeMaxDynamicSharedMemorySize, SMEM_TOT);
  attn_kernel<<<dim3(S_ / 16, B_ * H_), 256, SMEM_TOT, stream>>>(
      Qb, Ktb, Vb, Wepi, bo, out_eff, out_attn);
}
